// SelfAttention_24893630447757
// MI455X (gfx1250) — matrix operand order repaired and checked by reading, not compiled, not run on hardware
//
#include <hip/hip_runtime.h>

typedef __attribute__((ext_vector_type(16))) __bf16 v16bf;
typedef __attribute__((ext_vector_type(8)))  __bf16 v8bf;
typedef __attribute__((ext_vector_type(8)))  float  v8f;

#define EPS_BN 1e-5f
#define EPS_RENORM 1e-9f

// ---------------- async global->LDS (gfx1250) with graceful fallback ----------------
#if __has_builtin(__builtin_amdgcn_global_load_async_to_lds_b128)
#define HAS_ASYNC 1
typedef int v4i_vs __attribute__((vector_size(16)));
typedef __attribute__((address_space(1))) v4i_vs gv4i_t;   // global 128-bit payload
typedef __attribute__((address_space(3))) v4i_vs lv4i_t;   // LDS 128-bit payload
#else
#define HAS_ASYNC 0
#endif

__device__ __forceinline__ void async_cp16(const __bf16* g, __bf16* l) {
#if HAS_ASYNC
  __builtin_amdgcn_global_load_async_to_lds_b128((gv4i_t*)g, (lv4i_t*)l, 0, 0);
#else
  *(v8bf*)l = *(const v8bf*)g;
#endif
}

__device__ __forceinline__ void async_wait0() {
#if HAS_ASYNC
#if __has_builtin(__builtin_amdgcn_s_wait_asynccnt)
  __builtin_amdgcn_s_wait_asynccnt(0);
#else
  asm volatile("s_wait_asynccnt 0x0" ::: "memory");
#endif
#endif
}

// ---------- fold BN into weights: W'[o][i] = W*scale[o] (bf16), bias[o] = b - m*scale ----------
__global__ void fold_bn_kernel(const float* __restrict__ W, const float* __restrict__ g,
                               const float* __restrict__ b, const float* __restrict__ m,
                               const float* __restrict__ v,
                               __bf16* __restrict__ Wout, float* __restrict__ biasOut,
                               int Cout, int Cin) {
  int i = blockIdx.x * blockDim.x + threadIdx.x;
  if (i < Cout * Cin) {
    int o = i / Cin;
    float s = g[o] * rsqrtf(v[o] + EPS_BN);
    Wout[i] = (__bf16)(W[i] * s);
  }
  if (i < Cout) {
    float s = g[i] * rsqrtf(v[i] + EPS_BN);
    biasOut[i] = b[i] - m[i] * s;
  }
}

// ---------- x [b][c][n] fp32 -> xT [b][n][c] bf16 (LDS-tiled transpose) ----------
__global__ __launch_bounds__(256)
void transpose_cvt_kernel(const float* __restrict__ x, __bf16* __restrict__ xT, int C, int N) {
  __shared__ __bf16 tile[32][33];
  int n0 = blockIdx.x * 32, c0 = blockIdx.y * 32;
  const float* xb = x + (long long)blockIdx.z * C * N;
  __bf16* xTb = xT + (long long)blockIdx.z * C * N;
  int tx = threadIdx.x & 31, ty = threadIdx.x >> 5;  // 32 x 8
  #pragma unroll
  for (int i = 0; i < 32; i += 8)
    tile[ty + i][tx] = (__bf16)xb[(long long)(c0 + ty + i) * N + (n0 + tx)];
  __syncthreads();
  #pragma unroll
  for (int i = 0; i < 32; i += 8)
    xTb[(long long)(n0 + ty + i) * C + (c0 + tx)] = tile[tx][ty + i];
}

// ---------------- generic WMMA GEMM ----------------
// C[M x N] = A[M x K] (bf16 row-major) * B
//   BTRANS=false: B given K-contiguous as Bt[n][k] (ldb = K-leading) -> pure async row-chunk staging
//   BTRANS=true : B given row-major [k][n] (ldb = N-leading) -> manual transposed staging
// MODE 1: out bf16[m*ldc+n]  = relu(acc + bias[m])
// MODE 5: out bf16[n*ldc+m]  = relu(acc + bias[m])           (transposed, packed stores)
// MODE 2: out bf16[m*ldc+n]  = acc                            (raw scores)
// MODE 3: out bf16[n*ldc+m]  = x1T[n*ldAux+m] - acc*recip[n]  (transposed, packed stores)
// MODE 4: out fp32[m*ldc+n]  = x1T[n*ldAux+m] + relu(acc + bias[m])
template<int MODE, bool BTRANS>
__global__ __launch_bounds__(256)
void gemm_wmma_kernel(const __bf16* __restrict__ A, const __bf16* __restrict__ B,
                      void* __restrict__ outp,
                      const float* __restrict__ bias,
                      const __bf16* __restrict__ aux,     // x1T
                      const float* __restrict__ recip,    // 1/(eps+colsum)
                      int K, int lda, int ldb, int ldc, int ldAux,
                      long long sA, long long sB, long long sC,
                      long long sAux, long long sRecip) {
  constexpr int BM = 128, BN = 128, BK = 32, LDST = BK + 8;
  __shared__ __bf16 As[2][BM * LDST];
  __shared__ __bf16 Bs[2][BN * LDST];   // always [n][k] in LDS

  const int tid  = threadIdx.x;
  const int wave = tid >> 5;
  const int lane = tid & 31;
  const int half = lane >> 4;
  const int l16  = lane & 15;
  const int wm = (wave >> 2) * 64;
  const int wn = (wave & 3) * 32;
  const int m0 = blockIdx.y * BM;
  const int n0 = blockIdx.x * BN;
  const long long z = blockIdx.z;

  A += z * sA;
  B += z * sB;

  auto stageA = [&](int buf, int k0) {
    for (int c = tid; c < (BM * BK) / 8; c += 256) {
      int row = c >> 2, kc = (c & 3) << 3;
      async_cp16(A + (long long)(m0 + row) * lda + (k0 + kc), &As[buf][row * LDST + kc]);
    }
  };
  auto stageB = [&](int buf, int k0) {
    if constexpr (BTRANS) {   // B row-major [k][n]: load chunks, scatter-transpose into LDS
      for (int c = tid; c < (BK * BN) / 8; c += 256) {
        int kk = c >> 4, nc = (c & 15) << 3;
        v8bf bv = *(const v8bf*)(B + (long long)(k0 + kk) * ldb + (n0 + nc));
        #pragma unroll
        for (int e = 0; e < 8; e++) Bs[buf][(nc + e) * LDST + kk] = bv[e];
      }
    } else {                  // Bt[n][k]: identical pattern to A staging (async)
      for (int c = tid; c < (BN * BK) / 8; c += 256) {
        int row = c >> 2, kc = (c & 3) << 3;
        async_cp16(B + (long long)(n0 + row) * ldb + (k0 + kc), &Bs[buf][row * LDST + kc]);
      }
    }
  };

  v8f acc[4][2];
  #pragma unroll
  for (int i = 0; i < 4; i++)
    #pragma unroll
    for (int j = 0; j < 2; j++)
      #pragma unroll
      for (int e = 0; e < 8; e++) acc[i][j][e] = 0.0f;

  stageA(0, 0);
  stageB(0, 0);
  int p = 0;
  for (int k0 = 0; k0 < K; k0 += BK) {
    async_wait0();
    __syncthreads();                       // buf p fully staged for all waves
    if (k0 + BK < K) { stageA(p ^ 1, k0 + BK); stageB(p ^ 1, k0 + BK); }  // prefetch

    const __bf16* Ab = &As[p][0];
    const __bf16* Bb = &Bs[p][0];
    // A 16x32 frag: lane<16 -> row m=l16, K {0..7,16..23}; lane>=16 -> K {8..15,24..31}
    v16bf af[4];
    #pragma unroll
    for (int i = 0; i < 4; i++) {
      const __bf16* pa = Ab + (wm + i * 16 + l16) * LDST + half * 8;
      v8bf lo = *(const v8bf*)pa;
      v8bf hi = *(const v8bf*)(pa + 16);
      #pragma unroll
      for (int e = 0; e < 8; e++) { af[i][e] = lo[e]; af[i][e + 8] = hi[e]; }
    }
    // B 32x16 frag: lane<16 -> col n=l16, K {0..7,16..23}; lane>=16 -> K {8..15,24..31}
    v16bf bq[2];
    #pragma unroll
    for (int j = 0; j < 2; j++) {
      const __bf16* pb = Bb + (wn + j * 16 + l16) * LDST + half * 8;
      v8bf lo = *(const v8bf*)pb;
      v8bf hi = *(const v8bf*)(pb + 16);
      #pragma unroll
      for (int e = 0; e < 8; e++) { bq[j][e] = lo[e]; bq[j][e + 8] = hi[e]; }
    }

    #pragma unroll
    for (int i = 0; i < 4; i++)
      #pragma unroll
      for (int j = 0; j < 2; j++)
        acc[i][j] = __builtin_amdgcn_wmma_f32_16x16x32_bf16(
            false, af[i], false, bq[j], (short)0, acc[i][j], false, false);

    __syncthreads();                       // all waves done reading buf p
    p ^= 1;
  }

  // ---- epilogue ----
  #pragma unroll
  for (int i = 0; i < 4; i++) {
    int mb = m0 + wm + i * 16 + half * 8;
    #pragma unroll
    for (int j = 0; j < 2; j++) {
      int n = n0 + wn + j * 16 + l16;
      if constexpr (MODE == 1) {
        __bf16* out = (__bf16*)outp + z * sC;
        #pragma unroll
        for (int g = 0; g < 8; g++)
          out[(long long)(mb + g) * ldc + n] = (__bf16)fmaxf(acc[i][j][g] + bias[mb + g], 0.0f);
      } else if constexpr (MODE == 5) {
        __bf16* out = (__bf16*)outp + z * sC;
        v8bf pack;
        #pragma unroll
        for (int g = 0; g < 8; g++)
          pack[g] = (__bf16)fmaxf(acc[i][j][g] + bias[mb + g], 0.0f);
        *(v8bf*)(out + (long long)n * ldc + mb) = pack;
      } else if constexpr (MODE == 2) {
        __bf16* out = (__bf16*)outp + z * sC;
        #pragma unroll
        for (int g = 0; g < 8; g++)
          out[(long long)(mb + g) * ldc + n] = (__bf16)acc[i][j][g];
      } else if constexpr (MODE == 3) {
        __bf16* out = (__bf16*)outp + z * sC;
        const __bf16* x1p = aux + z * sAux + (long long)n * ldAux + mb;
        float rc = recip[z * sRecip + n];
        v8bf xv = *(const v8bf*)x1p;
        v8bf pack;
        #pragma unroll
        for (int g = 0; g < 8; g++)
          pack[g] = (__bf16)((float)xv[g] - acc[i][j][g] * rc);
        *(v8bf*)(out + (long long)n * ldc + mb) = pack;
      } else { // MODE 4
        float* out = (float*)outp + z * sC;
        const __bf16* x1p = aux + z * sAux + (long long)n * ldAux + mb;
        v8bf xv = *(const v8bf*)x1p;
        #pragma unroll
        for (int g = 0; g < 8; g++)
          out[(long long)(mb + g) * ldc + n] =
              (float)xv[g] + fmaxf(acc[i][j][g] + bias[mb + g], 0.0f);
      }
    }
  }
}

// ---------- row softmax in place on bf16 scores (one block = one row of 2048) ----------
__global__ __launch_bounds__(256)
void softmax_rows_kernel(__bf16* __restrict__ S, int N) {
  __bf16* row = S + (long long)blockIdx.x * N;
  int tid = threadIdx.x;
  float vals[8];
  float mx = -1e30f;
  #pragma unroll
  for (int i = 0; i < 8; i++) { vals[i] = (float)row[tid + i * 256]; mx = fmaxf(mx, vals[i]); }
  __shared__ float red[256];
  red[tid] = mx; __syncthreads();
  for (int s = 128; s > 0; s >>= 1) { if (tid < s) red[tid] = fmaxf(red[tid], red[tid + s]); __syncthreads(); }
  mx = red[0]; __syncthreads();
  float sum = 0.0f;
  #pragma unroll
  for (int i = 0; i < 8; i++) { vals[i] = __expf(vals[i] - mx); sum += vals[i]; }
  red[tid] = sum; __syncthreads();
  for (int s = 128; s > 0; s >>= 1) { if (tid < s) red[tid] += red[tid + s]; __syncthreads(); }
  float inv = 1.0f / red[0];
  #pragma unroll
  for (int i = 0; i < 8; i++) row[tid + i * 256] = (__bf16)(vals[i] * inv);
}

// ---------- recip[b][m] = 1/(eps + sum_n S[b][n][m]) ----------
__global__ void colsum_kernel(const __bf16* __restrict__ S, float* __restrict__ recip, int N) {
  int t = blockIdx.x * blockDim.x + threadIdx.x;
  int b = t / N, mcol = t % N;
  const __bf16* p = S + (long long)b * N * N + mcol;
  float s = 0.0f;
  for (int n = 0; n < N; n++) s += (float)p[(long long)n * N];
  recip[t] = 1.0f / (EPS_RENORM + s);
}

extern "C" void kernel_launch(void* const* d_in, const int* in_sizes, int n_in,
                              void* d_out, int out_size, void* d_ws, size_t ws_size,
                              hipStream_t stream) {
  const int Bb = 8, C = 1024, N = 2048, Cq = 128;
  const float* x = (const float*)d_in[0];
  const float *W[5], *g[5], *bb[5], *mm[5], *vv[5];
  for (int i = 0; i < 5; i++) {
    W[i]  = (const float*)d_in[1 + 5 * i];
    g[i]  = (const float*)d_in[2 + 5 * i];
    bb[i] = (const float*)d_in[3 + 5 * i];
    mm[i] = (const float*)d_in[4 + 5 * i];
    vv[i] = (const float*)d_in[5 + 5 * i];
  }

  char* ws = (char*)d_ws;
  size_t off = 0;
  auto alloc = [&](size_t bytes) { size_t o = off; off = (off + bytes + 255) & ~(size_t)255; return o; };
  const int cout_[5] = {1024, 1024, 128, 128, 1024};
  __bf16* Wf[5]; float* biasf[5];
  for (int i = 0; i < 5; i++) {
    Wf[i]    = (__bf16*)(ws + alloc((size_t)cout_[i] * 1024 * 2));
    biasf[i] = (float*)(ws + alloc(1024 * 4));
  }
  __bf16* xT   = (__bf16*)(ws + alloc((size_t)Bb * C * N * 2));   // [b][n][c]
  __bf16* x1T  = (__bf16*)(ws + alloc((size_t)Bb * C * N * 2));   // [b][n][c]
  __bf16* vbuf = (__bf16*)(ws + alloc((size_t)Bb * C * N * 2));   // [b][c][n]
  __bf16* qT   = (__bf16*)(ws + alloc((size_t)Bb * N * Cq * 2));  // [b][n][128]
  __bf16* kT   = (__bf16*)(ws + alloc((size_t)Bb * N * Cq * 2));  // [b][m][128]
  __bf16* Sb   = (__bf16*)(ws + alloc((size_t)Bb * N * N * 2));   // [b][n][m]
  __bf16* tT   = (__bf16*)(ws + alloc((size_t)Bb * C * N * 2));   // [b][n][c]
  float*  recip = (float*)(ws + alloc((size_t)Bb * N * 4));

  for (int i = 0; i < 5; i++) {
    int total = cout_[i] * 1024;
    fold_bn_kernel<<<(total + 255) / 256, 256, 0, stream>>>(
        W[i], g[i], bb[i], mm[i], vv[i], Wf[i], biasf[i], cout_[i], 1024);
  }
  transpose_cvt_kernel<<<dim3(N / 32, C / 32, Bb), 256, 0, stream>>>(x, xT, C, N);

  dim3 blk(256);
  const long long sBN = (long long)C * N;
  const long long sQK = (long long)Cq * N;
  const long long sS  = (long long)N * N;

  // x1T[n][c] = relu(BN(W1 x))
  gemm_wmma_kernel<5, false><<<dim3(N / 128, C / 128, Bb), blk, 0, stream>>>(
      Wf[0], xT, x1T, biasf[0], nullptr, nullptr, 1024, 1024, 1024, 1024, 0, 0, sBN, sBN, 0, 0);
  // v[c][n] = relu(BN(W2 x1))
  gemm_wmma_kernel<1, false><<<dim3(N / 128, C / 128, Bb), blk, 0, stream>>>(
      Wf[1], x1T, vbuf, biasf[1], nullptr, nullptr, 1024, 1024, 1024, N, 0, 0, sBN, sBN, 0, 0);
  // qT[n][c'] = relu(BN(W3 x1))
  gemm_wmma_kernel<5, false><<<dim3(N / 128, Cq / 128, Bb), blk, 0, stream>>>(
      Wf[2], x1T, qT, biasf[2], nullptr, nullptr, 1024, 1024, 1024, Cq, 0, 0, sBN, sQK, 0, 0);
  // kT[m][c'] = relu(BN(W4 x1))
  gemm_wmma_kernel<5, false><<<dim3(N / 128, Cq / 128, Bb), blk, 0, stream>>>(
      Wf[3], x1T, kT, biasf[3], nullptr, nullptr, 1024, 1024, 1024, Cq, 0, 0, sBN, sQK, 0, 0);
  // Sb[n][m] = qT @ kT^T  (K=128), raw bf16
  gemm_wmma_kernel<2, false><<<dim3(N / 128, N / 128, Bb), blk, 0, stream>>>(
      qT, kT, Sb, nullptr, nullptr, nullptr, Cq, Cq, Cq, N, 0, sQK, sQK, sS, 0, 0);
  // softmax along rows, in place (bf16)
  softmax_rows_kernel<<<dim3(Bb * N), blk, 0, stream>>>(Sb, N);
  // recip[b][m]
  colsum_kernel<<<dim3(Bb * N / 256), blk, 0, stream>>>(Sb, recip, N);
  // tT[m'][c] = x1T - (v @ S)*recip   (B = Sb row-major -> manual transpose staging)
  gemm_wmma_kernel<3, true><<<dim3(N / 128, C / 128, Bb), blk, 0, stream>>>(
      vbuf, Sb, tT, nullptr, x1T, recip, N, N, N, 1024, 1024, sBN, sS, sBN, sBN, N);
  // out[c][n] fp32 = x1 + relu(BN(W5 t))
  gemm_wmma_kernel<4, false><<<dim3(N / 128, C / 128, Bb), blk, 0, stream>>>(
      Wf[4], tT, (float*)d_out, biasf[4], x1T, nullptr, 1024, 1024, 1024, N, 1024, 0, sBN, sBN, sBN, 0);
}
